// BiLSTM_18700287607567
// MI455X (gfx1250) — compile-verified, matrix operand order checked by reading, not run on hardware
//
#include <hip/hip_runtime.h>
#include <hip/hip_bf16.h>
#include <cstdint>

// ---------------------------------------------------------------------------
// Types for CDNA5 WMMA (wave32): A/B = v16h (16 halves/lane), C/D = v8f.
// ---------------------------------------------------------------------------
typedef __attribute__((ext_vector_type(16))) _Float16 v16h;
typedef __attribute__((ext_vector_type(8)))  _Float16 v8h;
typedef __attribute__((ext_vector_type(8)))  float    v8f;

__device__ __forceinline__ v8f wmma_f16(v16h a, v16h b, v8f c) {
    // D = A(16x32) * B(32x16) + C, f32 accumulate
    return __builtin_amdgcn_wmma_f32_16x16x32_f16(
        /*neg_a=*/false, a, /*neg_b=*/false, b,
        /*c_mod=*/(short)0, c, /*reuse_a=*/false, /*reuse_b=*/false);
}

// ---------------------------------------------------------------------------
// Async global->LDS 16B copy (ASYNCcnt path, CDNA5 ISA).
// Builtin signature (from clang diagnostic): (v4i AS1*, v4i AS3*, imm, imm).
// ---------------------------------------------------------------------------
typedef int v4i __attribute__((vector_size(16)));
typedef __attribute__((address_space(1))) v4i gv4i_t;
typedef __attribute__((address_space(3))) v4i lv4i_t;

__device__ __forceinline__ void async_ld_b128(const _Float16* g, _Float16* l) {
#if defined(__AMDGCN__)
    gv4i_t* gp = (gv4i_t*)(unsigned long long)(uintptr_t)g;   // global address
    lv4i_t* lp = (lv4i_t*)(unsigned)(uintptr_t)l;             // low 32b = LDS offset
  #if __has_builtin(__builtin_amdgcn_global_load_async_to_lds_b128)
    __builtin_amdgcn_global_load_async_to_lds_b128(gp, lp, 0, 0);
  #else
    asm volatile("global_load_async_to_lds_b128 %0, %1, off"
                 :: "v"(lp), "v"(gp) : "memory");
  #endif
#endif
}
__device__ __forceinline__ void wait_async_0() {
#if defined(__AMDGCN__)
  #if __has_builtin(__builtin_amdgcn_s_wait_asynccnt)
    __builtin_amdgcn_s_wait_asynccnt(0);
  #else
    asm volatile("s_wait_asynccnt 0x0" ::: "memory");
  #endif
#endif
}
__device__ __forceinline__ void wait_async_2() {
#if defined(__AMDGCN__)
  #if __has_builtin(__builtin_amdgcn_s_wait_asynccnt)
    __builtin_amdgcn_s_wait_asynccnt(2);
  #else
    asm volatile("s_wait_asynccnt 0x2" ::: "memory");
  #endif
#endif
}

// Load one 16x32 (f16) A- or B-fragment from LDS.
// ISA layout (CDNA5 ISA §7.12.2, 16-bit 16x32): lane L holds row r = L&15;
// lanes 0-15 hold K = {0..7, 16..23}, lanes 16-31 hold K = {8..15, 24..31}.
__device__ __forceinline__ v16h load_frag(const _Float16* base, int stride, int k0) {
    const int lane = threadIdx.x & 31;
    const int r    = lane & 15;
    const int kh   = (lane >> 4) << 3;          // 0 or 8
    const _Float16* p = base + r * stride + k0 + kh;
    v8h lo = *(const v8h*)(p);                  // ds_read_b128
    v8h hi = *(const v8h*)(p + 16);             // ds_read_b128
    return __builtin_shufflevector(lo, hi, 0,1,2,3,4,5,6,7,8,9,10,11,12,13,14,15);
}

__device__ __forceinline__ float sigm(float x)  { return 1.0f / (1.0f + __expf(-x)); }
__device__ __forceinline__ float ftanh(float x) { return 2.0f * sigm(2.0f * x) - 1.0f; }

// Scan-layout flat index (f16): block (te,mb) = dense 16KB chunk; inside:
// gate q (stride 2048), wave w (256), lane L (8), g — the exact accumulator
// ordering lstm_scan needs, so pre reads are 4 contiguous b128 per lane.
__device__ __forceinline__ size_t scan_idx(int te, int mb, int q, int w, int L, int g) {
    return ((((size_t)te * 4 + mb) * 4 + q) * 8 + w) * 256 + (size_t)L * 8 + g;
}

// ---------------------------------------------------------------------------
// f32 -> f16 conversion (one-time, for x / w_ih / w_hh / w_fc)
// ---------------------------------------------------------------------------
__global__ void cvt_f32_f16(const float* __restrict__ src, _Float16* __restrict__ dst, int n) {
    int i = blockIdx.x * blockDim.x + threadIdx.x;
    if (i < n) dst[i] = (_Float16)src[i];
}

// ---------------------------------------------------------------------------
// GEMM:  C(MxN) = A(MxK,f16) * W(NxK,f16)^T + bias0 + bias1
// Async-pipelined: 3-slot LDS rotation; per chunk: issue async c+1,
// s_wait_asynccnt 2, one barrier, wmma on chunk c. Depth-3 guarantees a
// barrier separates a slot's last read from its async rewrite.
// Block = 256 threads (8 waves), 64x64 tile; wave w owns m-tile (w&3) and
// n-tiles {2*(w>>2), 2*(w>>2)+1}; K consumed in 32-wide chunks.
// SCAN=true: N==512, C f16 in scan layout. SCAN=false: C row-major.
// ---------------------------------------------------------------------------
template <typename OT, bool SCAN>
__global__ void gemm_f16(const _Float16* __restrict__ A, const _Float16* __restrict__ W,
                         const float* __restrict__ bias0, const float* __restrict__ bias1,
                         OT* __restrict__ C, int M, int N, int K)
{
    __shared__ _Float16 As[3][64][40];   // stride 40 halves (80B, 16B multiple)
    __shared__ _Float16 Bs[3][64][40];

    const int tid  = threadIdx.x;
    const int bm   = blockIdx.y * 64;
    const int bn   = blockIdx.x * 64;
    const int wave = tid >> 5;
    const int lane = tid & 31;
    const int mt   = wave & 3;
    const int nt0  = (wave >> 2) << 1;

    const int srow = tid >> 2;           // 0..63
    const int sseg = (tid & 3) * 8;      // 0,8,16,24

    v8f acc0 = {}, acc1 = {};

    const int NC = K >> 5;
    // prologue: chunk 0 -> slot 0
    async_ld_b128(A + (size_t)(bm + srow) * K + sseg, &As[0][srow][sseg]);
    async_ld_b128(W + (size_t)(bn + srow) * K + sseg, &Bs[0][srow][sseg]);

    for (int c = 0; c < NC; ++c) {
        if (c + 1 < NC) {
            const int s1 = (c + 1) % 3, kc = (c + 1) << 5;
            async_ld_b128(A + (size_t)(bm + srow) * K + kc + sseg, &As[s1][srow][sseg]);
            async_ld_b128(W + (size_t)(bn + srow) * K + kc + sseg, &Bs[s1][srow][sseg]);
            wait_async_2();              // chunk c landed (2 newest still in flight)
        } else {
            wait_async_0();
        }
        __syncthreads();
        const int s = c % 3;
        v16h af  = load_frag(&As[s][mt * 16][0], 40, 0);
        v16h bf0 = load_frag(&Bs[s][nt0 * 16][0], 40, 0);
        acc0 = wmma_f16(af, bf0, acc0);
        v16h bf1 = load_frag(&Bs[s][(nt0 + 1) * 16][0], 40, 0);
        acc1 = wmma_f16(af, bf1, acc1);
    }

    // D layout: lane L, VGPR g -> (m = g + 8*(L>>4), n = L&15)
    const int jl   = lane & 15;
    const int half = lane >> 4;
    #pragma unroll
    for (int q = 0; q < 2; ++q) {
        v8f acc = q ? acc1 : acc0;
        const int gn = bn + (nt0 + q) * 16 + jl;
        float bs = (bias0 ? bias0[gn] : 0.0f) + (bias1 ? bias1[gn] : 0.0f);
        #pragma unroll
        for (int g = 0; g < 8; ++g) {
            const int gm = bm + mt * 16 + g + 8 * half;   // = b*512 + te
            if constexpr (SCAN) {
                const int b   = gm >> 9,  te  = gm & 511;
                const int mbi = b >> 4,   ms  = b & 15;    // batch tile / row-in-tile
                const int qg  = gn >> 7;                   // gate 0..3
                const int rem = gn & 127;
                const int wv  = rem >> 4, jls = rem & 15;  // scan wave / lane-low
                const int Ls  = jls + ((ms >> 3) << 4);    // scan lane
                C[scan_idx(te, mbi, qg, wv, Ls, ms & 7)] = (OT)(acc[g] + bs);
            } else {
                C[(size_t)gm * N + gn] = (OT)(acc[g] + bs);
            }
        }
    }
}

// ---------------------------------------------------------------------------
// LSTM scan. grid = 8 blocks: dir = blockIdx.x>>2, batch tile = blockIdx.x&3.
// w_hh (f16) pulled into LDS with async b128 copies and kept resident
// (128 KB of the 320 KB WGP LDS); h ping-pongs between two LDS buffers
// (single barrier/step); c lives in registers for all 512 steps. Wave w
// computes gate columns {j, j+128, j+256, j+384}: i/f/g/o for one (m,j) land
// in the same lane+VGPR of acc0..acc3; pre read as 4 b128 loads per lane.
// ---------------------------------------------------------------------------
__global__ void lstm_scan(const _Float16* __restrict__ pre_f,
                          const _Float16* __restrict__ pre_b,
                          const _Float16* __restrict__ whh_f,
                          const _Float16* __restrict__ whh_b,
                          _Float16* __restrict__ h_out)
{
    const int dir = blockIdx.x >> 2;
    const int mb  = blockIdx.x & 3;
    const _Float16* __restrict__ pre = dir ? pre_b : pre_f;
    const _Float16* __restrict__ whh = dir ? whh_b : whh_f;

    __shared__ _Float16 Wl[512][136];     // Wl[n][k] (f16), 139,264 B, row 272B (16B mult)
    __shared__ _Float16 Hl[2][16][136];   // h state ping-pong, 8,704 B

    const int tid = threadIdx.x;
    // async preload of w_hh: 8192 x 16B chunks
    for (int it = tid; it < 8192; it += 256) {
        const int r = it >> 4, seg = (it & 15) << 3;
        async_ld_b128(whh + r * 128 + seg, &Wl[r][seg]);
    }
    for (int i = tid; i < 2 * 16 * 136; i += 256) (&Hl[0][0][0])[i] = (_Float16)0.0f;
    wait_async_0();
    __syncthreads();

    const int wave = tid >> 5;
    const int lane = tid & 31;
    const int jl   = lane & 15;
    const int half = lane >> 4;
    const int j    = wave * 16 + jl;   // hidden unit 0..127
    const int rowb = mb * 16;          // batch row base

    float creg[8];
    #pragma unroll
    for (int g = 0; g < 8; ++g) creg[g] = 0.0f;

    for (int t = 0; t < 512; ++t) {
        const int te = dir ? (511 - t) : t;
        const int rb = t & 1, wb = rb ^ 1;

        // issue this step's 4 pre b128 loads early; prefetch next 16KB chunk
        const v8h p0 = *(const v8h*)(pre + scan_idx(te, mb, 0, wave, lane, 0));
        const v8h p1 = *(const v8h*)(pre + scan_idx(te, mb, 1, wave, lane, 0));
        const v8h p2 = *(const v8h*)(pre + scan_idx(te, mb, 2, wave, lane, 0));
        const v8h p3 = *(const v8h*)(pre + scan_idx(te, mb, 3, wave, lane, 0));
        if (t + 1 < 512) {
            const int tn = dir ? (te - 1) : (te + 1);
            __builtin_prefetch(pre + ((size_t)tn * 4 + mb) * 8192 + (size_t)tid * 32, 0, 1);
        }

        v8f acc0 = {}, acc1 = {}, acc2 = {}, acc3 = {};
        #pragma unroll
        for (int kc = 0; kc < 128; kc += 32) {
            v16h af = load_frag(&Hl[rb][0][0], 136, kc);                // shared by all gates
            v16h b0 = load_frag(&Wl[0 * 128 + wave * 16][0], 136, kc);
            acc0 = wmma_f16(af, b0, acc0);
            v16h b1 = load_frag(&Wl[1 * 128 + wave * 16][0], 136, kc);
            acc1 = wmma_f16(af, b1, acc1);
            v16h b2 = load_frag(&Wl[2 * 128 + wave * 16][0], 136, kc);
            acc2 = wmma_f16(af, b2, acc2);
            v16h b3 = load_frag(&Wl[3 * 128 + wave * 16][0], 136, kc);
            acc3 = wmma_f16(af, b3, acc3);
        }

        float hnew[8];
        #pragma unroll
        for (int g = 0; g < 8; ++g) {
            const float iv = sigm (acc0[g] + (float)p0[g]);
            const float fv = sigm (acc1[g] + (float)p1[g]);
            const float gv = ftanh(acc2[g] + (float)p2[g]);
            const float ov = sigm (acc3[g] + (float)p3[g]);
            const float cn = fv * creg[g] + iv * gv;
            creg[g] = cn;
            hnew[g] = ov * ftanh(cn);
        }

        #pragma unroll
        for (int g = 0; g < 8; ++g) {
            const int m = g + 8 * half;
            const _Float16 hv = (_Float16)hnew[g];
            Hl[wb][m][j] = hv;   // other buffer: no read hazard this step
            h_out[((size_t)(rowb + m) * 512 + te) * 256 + dir * 128 + j] = hv;
        }
        __syncthreads();         // Hl[wb] visible before next step reads it
    }
}

// ---------------------------------------------------------------------------
// Launcher
// ---------------------------------------------------------------------------
extern "C" void kernel_launch(void* const* d_in, const int* in_sizes, int n_in,
                              void* d_out, int out_size, void* d_ws, size_t ws_size,
                              hipStream_t stream)
{
    const float* x        = (const float*)d_in[0];
    const float* w_ih_l0  = (const float*)d_in[1];
    const float* w_hh_l0  = (const float*)d_in[2];
    const float* b_ih_l0  = (const float*)d_in[3];
    const float* b_hh_l0  = (const float*)d_in[4];
    const float* w_ih_l0r = (const float*)d_in[5];
    const float* w_hh_l0r = (const float*)d_in[6];
    const float* b_ih_l0r = (const float*)d_in[7];
    const float* b_hh_l0r = (const float*)d_in[8];
    const float* w_ih_l1  = (const float*)d_in[9];
    const float* w_hh_l1  = (const float*)d_in[10];
    const float* b_ih_l1  = (const float*)d_in[11];
    const float* b_hh_l1  = (const float*)d_in[12];
    const float* w_ih_l1r = (const float*)d_in[13];
    const float* w_hh_l1r = (const float*)d_in[14];
    const float* b_ih_l1r = (const float*)d_in[15];
    const float* b_hh_l1r = (const float*)d_in[16];
    const float* w_fc     = (const float*)d_in[17];
    const float* b_fc     = (const float*)d_in[18];

    const size_t MB = 1024 * 1024;
    char* ws = (char*)d_ws;
    _Float16* pre_fw = (_Float16*)(ws);                 // 32 MB (scan layout)
    _Float16* pre_bw = (_Float16*)(ws + 32 * MB);       // 32 MB
    _Float16* h0     = (_Float16*)(ws + 64 * MB);       // 16 MB
    _Float16* h1     = (_Float16*)(ws + 80 * MB);       // 16 MB
    _Float16* xh     = (_Float16*)(ws + 96 * MB);       // 4 MB  (x as f16)
    char* wbase      = ws + 100 * MB;
    _Float16* wih0f  = (_Float16*)(wbase);              //  64 KB
    _Float16* wih0b  = (_Float16*)(wbase + 64  * 1024);
    _Float16* wih1f  = (_Float16*)(wbase + 128 * 1024); // 256 KB
    _Float16* wih1b  = (_Float16*)(wbase + 384 * 1024);
    _Float16* wfch   = (_Float16*)(wbase + 640 * 1024); //  32 KB
    _Float16* whh0f  = (_Float16*)(wbase + 672 * 1024); // 128 KB each
    _Float16* whh0b  = (_Float16*)(wbase + 800 * 1024);
    _Float16* whh1f  = (_Float16*)(wbase + 928 * 1024);
    _Float16* whh1b  = (_Float16*)(wbase + 1056 * 1024);

    const int M = 64 * 512;            // 32768 time-batch rows
    dim3 blk(256);
    auto cvt = [&](const float* s, _Float16* d, int n) {
        cvt_f32_f16<<<dim3((n + 255) / 256), blk, 0, stream>>>(s, d, n);
    };

    // ---- one-time f32 -> f16 conversions ----
    cvt(x,        xh,    M * 64);
    cvt(w_ih_l0,  wih0f, 512 * 64);
    cvt(w_ih_l0r, wih0b, 512 * 64);
    cvt(w_ih_l1,  wih1f, 512 * 256);
    cvt(w_ih_l1r, wih1b, 512 * 256);
    cvt(w_fc,     wfch,  64 * 256);
    cvt(w_hh_l0,  whh0f, 512 * 128);
    cvt(w_hh_l0r, whh0b, 512 * 128);
    cvt(w_hh_l1,  whh1f, 512 * 128);
    cvt(w_hh_l1r, whh1b, 512 * 128);

    // ---- Layer 0: input projections (K=64), scan ----
    gemm_f16<_Float16, true><<<dim3(8, 512), blk, 0, stream>>>(
        xh, wih0f, b_ih_l0,  b_hh_l0,  pre_fw, M, 512, 64);
    gemm_f16<_Float16, true><<<dim3(8, 512), blk, 0, stream>>>(
        xh, wih0b, b_ih_l0r, b_hh_l0r, pre_bw, M, 512, 64);
    lstm_scan<<<dim3(8), blk, 0, stream>>>(pre_fw, pre_bw, whh0f, whh0b, h0);

    // ---- Layer 1: input projections (K=256), scan ----
    gemm_f16<_Float16, true><<<dim3(8, 512), blk, 0, stream>>>(
        h0, wih1f, b_ih_l1,  b_hh_l1,  pre_fw, M, 512, 256);
    gemm_f16<_Float16, true><<<dim3(8, 512), blk, 0, stream>>>(
        h0, wih1b, b_ih_l1r, b_hh_l1r, pre_bw, M, 512, 256);
    lstm_scan<<<dim3(8), blk, 0, stream>>>(pre_fw, pre_bw, whh1f, whh1b, h1);

    // ---- Final FC: (32768x256) x (256x64) + b_fc -> f32 out ----
    gemm_f16<float, false><<<dim3(1, 512), blk, 0, stream>>>(
        h1, wfch, b_fc, nullptr, (float*)d_out, M, 64, 256);
}
